// CopyMechanismMixin_13022340842315
// MI455X (gfx1250) — compile-verified, matrix operand order checked by reading, not run on hardware
//
#include <hip/hip_runtime.h>
#include <hip/hip_bf16.h>

// Problem constants (from reference):
#define B_ 4
#define T_ 256
#define M_ 512
#define D_ 1024
#define V_ 50257

typedef __attribute__((ext_vector_type(16))) __bf16 v16bf;
typedef __attribute__((ext_vector_type(8)))  float  v8f;
typedef unsigned short u16;
typedef unsigned int   u32;

union Frag16 { v16bf v; uint4 q[2]; };

// fp32 -> bf16, round-to-nearest-even
__device__ __forceinline__ u16 f2bf(float f) {
  u32 u = __builtin_bit_cast(u32, f);
  u32 r = u + 0x7FFFu + ((u >> 16) & 1u);
  return (u16)(r >> 16);
}

// ---------------------------------------------------------------------------
// Generic WMMA GEMM:  C[M,N] = A[M,K] * B[N,K]^T  (+bias[n]) (+tanh) per batch
// A row-major (lda), B row-major (ldb, i.e. stored as N x K), C row-major (ldc)
// Block = 256 threads = 8 waves (2x4); wave tile = 32x32 (2x2 WMMA frags);
// block tile = 64 rows x 128 cols. Requires M % 64 == 0, K % 32 == 0.
// N may be ragged: B-fragment row index is CLAMPED (not guarded) — garbage
// columns only feed output columns that the guarded epilogue never stores,
// so the hot K-loop stays branch-free (no exec-mask churn, no zero fills).
// EPI: 0=f32, 1=f32+bias, 2=f32+bias+tanh, 3=bf16+bias, 4=bf16
// ---------------------------------------------------------------------------
template<int EPI>
__global__ __launch_bounds__(256)
void gemm_bf16_wmma(const u16* __restrict__ A, int lda, long strideA,
                    const u16* __restrict__ Bm, int ldb, long strideB,
                    void* __restrict__ Cout, int ldc, long strideC,
                    int Mrows, int Ncols, int K,
                    const float* __restrict__ bias)
{
  const int bz = blockIdx.z;
  A  += (size_t)bz * strideA;
  Bm += (size_t)bz * strideB;
  const size_t cbase = (size_t)bz * strideC;

  const int lane = threadIdx.x & 31;
  const int wave = threadIdx.x >> 5;
  const int wr = wave >> 2;        // 0..1  (wave row)
  const int wc = wave & 3;         // 0..3  (wave col)
  const int tileM = blockIdx.y * 64  + wr * 32;
  const int tileN = blockIdx.x * 128 + wc * 32;

  const int r = lane & 15;         // row within A 16-frag / col within B-frag
  const int kh = (lane >> 4) * 8;  // lanes 16-31 hold the K+8 half
  const int h  = lane >> 4;

  // Per-lane base pointers, hoisted out of the K loop.
  const u16* ap0 = A + (size_t)(tileM + r) * lda + kh;
  const u16* ap1 = A + (size_t)(tileM + 16 + r) * lda + kh;
  const int bn0 = min(tileN + r, Ncols - 1);        // clamped B rows
  const int bn1 = min(tileN + 16 + r, Ncols - 1);
  const u16* bp0 = Bm + (size_t)bn0 * ldb + kh;
  const u16* bp1 = Bm + (size_t)bn1 * ldb + kh;

  v8f acc[2][2] = {};

  for (int k0 = 0; k0 < K; k0 += 32) {
    Frag16 a0, a1, b0, b1;
    a0.q[0] = *(const uint4*)(ap0);      a0.q[1] = *(const uint4*)(ap0 + 16);
    a1.q[0] = *(const uint4*)(ap1);      a1.q[1] = *(const uint4*)(ap1 + 16);
    b0.q[0] = *(const uint4*)(bp0);      b0.q[1] = *(const uint4*)(bp0 + 16);
    b1.q[0] = *(const uint4*)(bp1);      b1.q[1] = *(const uint4*)(bp1 + 16);
    __builtin_prefetch(bp0 + 32, 0, 1);  // global_prefetch_b8, stream B ahead
    __builtin_prefetch(bp1 + 32, 0, 1);
    ap0 += 32; ap1 += 32; bp0 += 32; bp1 += 32;

    acc[0][0] = __builtin_amdgcn_wmma_f32_16x16x32_bf16(
        false, a0.v, false, b0.v, (short)0, acc[0][0], false, false);
    acc[0][1] = __builtin_amdgcn_wmma_f32_16x16x32_bf16(
        false, a0.v, false, b1.v, (short)0, acc[0][1], false, false);
    acc[1][0] = __builtin_amdgcn_wmma_f32_16x16x32_bf16(
        false, a1.v, false, b0.v, (short)0, acc[1][0], false, false);
    acc[1][1] = __builtin_amdgcn_wmma_f32_16x16x32_bf16(
        false, a1.v, false, b1.v, (short)0, acc[1][1], false, false);
  }

  // Epilogue. C frag layout: lanes 0-15: (M=e, N=lane); lanes 16-31: (M=8+e, N=lane-16)
  #pragma unroll
  for (int i = 0; i < 2; ++i) {
    #pragma unroll
    for (int j = 0; j < 2; ++j) {
      const int col = tileN + j * 16 + r;
      if (col >= Ncols) continue;
      float bv = 0.0f;
      if (EPI == 1 || EPI == 2 || EPI == 3) bv = bias[col];
      #pragma unroll
      for (int e = 0; e < 8; ++e) {
        const int row = tileM + i * 16 + h * 8 + e;
        float v = acc[i][j][e] + bv;
        if (EPI == 2) v = tanhf(v);
        if (EPI == 3 || EPI == 4) {
          ((u16*)Cout)[cbase + (size_t)row * ldc + col] = f2bf(v);
        } else {
          ((float*)Cout)[cbase + (size_t)row * ldc + col] = v;
        }
      }
    }
  }
}

// ---------------------------------------------------------------------------
// Elementwise helpers
// ---------------------------------------------------------------------------
__global__ void k_f32_to_bf16(const float* __restrict__ in, u16* __restrict__ out, long n) {
  long i = (long)blockIdx.x * blockDim.x + threadIdx.x;
  if (i < n) out[i] = f2bf(in[i]);
}

// decoder (B,T,D) f32 -> dec_cat[:, :, 0:D] bf16 (ld 2D)
__global__ void k_pack_dec(const float* __restrict__ dec, u16* __restrict__ cat, long n) {
  long i = (long)blockIdx.x * blockDim.x + threadIdx.x;
  if (i >= n) return;
  long bt = i / D_, d = i % D_;
  cat[bt * (2 * D_) + d] = f2bf(dec[i]);
}

// memory_output (B,M,D) f32 -> memT (B,D,M) bf16
__global__ void k_transpose_mem(const float* __restrict__ mem, u16* __restrict__ memT, long n) {
  long i = (long)blockIdx.x * blockDim.x + threadIdx.x;
  if (i >= n) return;
  long b = i / ((long)D_ * M_);
  long rem = i % ((long)D_ * M_);
  long d = rem / M_, m = rem % M_;
  memT[i] = f2bf(mem[(b * M_ + m) * (long)D_ + d]);
}

// Masked softmax over rows of length M_ (in place); also emits bf16 copy.
__global__ __launch_bounds__(256)
void k_softmax_m(float* __restrict__ scores, u16* __restrict__ attn_bf,
                 const int* __restrict__ seqlen)
{
  const int t = blockIdx.x, b = blockIdx.y, tid = threadIdx.x;
  const size_t base = ((size_t)b * T_ + t) * M_;
  float* row = scores + base;
  const int L = seqlen[b];
  __shared__ float red[256];

  float mx = -INFINITY;
  for (int m = tid; m < M_; m += 256) {
    float x = (m < L) ? row[m] : -INFINITY;
    mx = fmaxf(mx, x);
  }
  red[tid] = mx; __syncthreads();
  for (int s = 128; s > 0; s >>= 1) { if (tid < s) red[tid] = fmaxf(red[tid], red[tid + s]); __syncthreads(); }
  mx = red[0]; __syncthreads();

  float sum = 0.0f;
  for (int m = tid; m < M_; m += 256)
    if (m < L) sum += expf(row[m] - mx);
  red[tid] = sum; __syncthreads();
  for (int s = 128; s > 0; s >>= 1) { if (tid < s) red[tid] += red[tid + s]; __syncthreads(); }
  const float inv = 1.0f / red[0];

  for (int m = tid; m < M_; m += 256) {
    float p = (m < L) ? expf(row[m] - mx) * inv : 0.0f;
    row[m] = p;
    attn_bf[base + m] = f2bf(p);
  }
}

// Fused finalize per (b,t):
//  p_gen = sigmoid(dot(dec_with_attn, W_gen) + b_gen)
//  online softmax over V of logits, scale by p_gen (in place on d_out)
//  scatter-add attn*(1-p_gen) at memory_ids
//  log() in place
__global__ __launch_bounds__(512)
void k_finalize(float* __restrict__ out, const float* __restrict__ dwa,
                const float* __restrict__ Wgen, const float* __restrict__ bgen,
                const float* __restrict__ attn, const int* __restrict__ mem_ids,
                const int* __restrict__ seqlen)
{
  const int t = blockIdx.x, b = blockIdx.y, tid = threadIdx.x;
  const size_t bt = (size_t)b * T_ + t;
  float* row = out + bt * V_;
  __shared__ float redA[512];
  __shared__ float redB[512];

  // p_gen
  const float* dv = dwa + bt * D_;
  float acc = 0.0f;
  for (int d = tid; d < D_; d += 512) acc += dv[d] * Wgen[d];
  redA[tid] = acc; __syncthreads();
  for (int s = 256; s > 0; s >>= 1) { if (tid < s) redA[tid] += redA[tid + s]; __syncthreads(); }
  const float pg = 1.0f / (1.0f + expf(-(redA[0] + bgen[0])));
  __syncthreads();

  // online softmax reduction over V (single read pass)
  float mx = -INFINITY, sm = 0.0f;
  for (int v = tid; v < V_; v += 512) {
    float x = row[v];
    if (x > mx) { sm = sm * expf(mx - x) + 1.0f; mx = x; }
    else        { sm += expf(x - mx); }
  }
  redA[tid] = mx; redB[tid] = sm; __syncthreads();
  for (int s = 256; s > 0; s >>= 1) {
    if (tid < s) {
      float m1 = redA[tid], s1 = redB[tid];
      float m2 = redA[tid + s], s2 = redB[tid + s];
      float Mx = fmaxf(m1, m2);
      redA[tid] = Mx;
      redB[tid] = s1 * expf(m1 - Mx) + s2 * expf(m2 - Mx);
    }
    __syncthreads();
  }
  mx = redA[0];
  const float scale = pg / redB[0];
  __syncthreads();

  for (int v = tid; v < V_; v += 512) row[v] = expf(row[v] - mx) * scale;
  __syncthreads();

  // scatter copy probabilities (duplicate ids -> atomic accumulate)
  const int L = seqlen[b];
  const float* arow = attn + bt * M_;
  const int* ids = mem_ids + (size_t)b * M_;
  const float cp = 1.0f - pg;
  for (int m = tid; m < L; m += 512) atomicAdd(&row[ids[m]], arow[m] * cp);
  __threadfence(); __syncthreads();

  for (int v = tid; v < V_; v += 512) row[v] = logf(row[v]);
}

// ---------------------------------------------------------------------------
// Launch
// ---------------------------------------------------------------------------
extern "C" void kernel_launch(void* const* d_in, const int* in_sizes, int n_in,
                              void* d_out, int out_size, void* d_ws, size_t ws_size,
                              hipStream_t stream)
{
  const float* dec     = (const float*)d_in[0];   // (B,T,D)
  const float* mem     = (const float*)d_in[1];   // (B,M,D)
  const int*   mseq    = (const int*)  d_in[2];   // (B,)
  const int*   mids    = (const int*)  d_in[3];   // (B,M)
  const float* W_copy  = (const float*)d_in[4];   // (D,D)
  const float* b_copy  = (const float*)d_in[5];   // (D,)
  const float* W_dec   = (const float*)d_in[6];   // (D,2D)
  const float* b_dec   = (const float*)d_in[7];   // (D,)
  const float* W_gen   = (const float*)d_in[8];   // (1,D)
  const float* b_gen   = (const float*)d_in[9];   // (1,)
  const float* W_out   = (const float*)d_in[10];  // (V,D)
  const float* b_out   = (const float*)d_in[11];  // (V,)
  float* out = (float*)d_out;                     // (B,T,V)

  // workspace layout (~127.2 MB total)
  char* base = (char*)d_ws;
  size_t off = 0;
  auto alloc = [&](size_t bytes) { char* p = base + off; off = (off + bytes + 255) & ~(size_t)255; return p; };
  u16*   wcopy_bf = (u16*)  alloc((size_t)D_ * D_ * 2);           //  2 MB
  u16*   wdec_bf  = (u16*)  alloc((size_t)D_ * 2 * D_ * 2);       //  4 MB
  u16*   wout_bf  = (u16*)  alloc((size_t)V_ * D_ * 2);           // 98 MB
  u16*   mem_bf   = (u16*)  alloc((size_t)B_ * M_ * D_ * 2);      //  4 MB
  u16*   memT_bf  = (u16*)  alloc((size_t)B_ * D_ * M_ * 2);      //  4 MB
  u16*   mproj_bf = (u16*)  alloc((size_t)B_ * M_ * D_ * 2);      //  4 MB
  u16*   dec_cat  = (u16*)  alloc((size_t)B_ * T_ * 2 * D_ * 2);  //  4 MB
  float* scores   = (float*)alloc((size_t)B_ * T_ * M_ * 4);      //  2 MB
  u16*   attn_bf  = (u16*)  alloc((size_t)B_ * T_ * M_ * 2);      //  1 MB
  float* dwa      = (float*)alloc((size_t)B_ * T_ * D_ * 4);      //  4 MB
  (void)ws_size; (void)in_sizes; (void)n_in; (void)out_size;

  auto blks = [](long n) { return (unsigned)((n + 255) / 256); };

  // bf16 staging
  k_f32_to_bf16<<<blks((long)D_ * D_), 256, 0, stream>>>(W_copy, wcopy_bf, (long)D_ * D_);
  k_f32_to_bf16<<<blks((long)D_ * 2 * D_), 256, 0, stream>>>(W_dec, wdec_bf, (long)D_ * 2 * D_);
  k_f32_to_bf16<<<blks((long)V_ * D_), 256, 0, stream>>>(W_out, wout_bf, (long)V_ * D_);
  k_f32_to_bf16<<<blks((long)B_ * M_ * D_), 256, 0, stream>>>(mem, mem_bf, (long)B_ * M_ * D_);
  k_pack_dec<<<blks((long)B_ * T_ * D_), 256, 0, stream>>>(dec, dec_cat, (long)B_ * T_ * D_);
  k_transpose_mem<<<blks((long)B_ * D_ * M_), 256, 0, stream>>>(mem, memT_bf, (long)B_ * D_ * M_);

  // 1) mem_proj[b] (M x D) = mem[b] (M,K=D) * W_copy^T  + b_copy -> bf16
  gemm_bf16_wmma<3><<<dim3(D_ / 128, M_ / 64, B_), 256, 0, stream>>>(
      mem_bf, D_, (long)M_ * D_, wcopy_bf, D_, 0,
      mproj_bf, D_, (long)M_ * D_, M_, D_, D_, b_copy);

  // 2) scores[b] (T x M) = dec[b] (T,K=D, lda=2D) * mem_proj[b]^T -> f32
  gemm_bf16_wmma<0><<<dim3(M_ / 128, T_ / 64, B_), 256, 0, stream>>>(
      dec_cat, 2 * D_, (long)T_ * 2 * D_, mproj_bf, D_, (long)M_ * D_,
      scores, M_, (long)T_ * M_, T_, M_, D_, nullptr);

  // 3) masked softmax over M; attn f32 in place + bf16 copy
  k_softmax_m<<<dim3(T_, B_), 256, 0, stream>>>(scores, attn_bf, mseq);

  // 4) attn_out[b] (T x D) = attn[b] (T,K=M) * memT[b]^T -> bf16 into dec_cat[:, D:2D]
  gemm_bf16_wmma<4><<<dim3(D_ / 128, T_ / 64, B_), 256, 0, stream>>>(
      attn_bf, M_, (long)T_ * M_, memT_bf, M_, (long)D_ * M_,
      dec_cat + D_, 2 * D_, (long)T_ * 2 * D_, T_, D_, M_, nullptr);

  // 5) dec_with_attn[b] (T x D) = tanh(dec_cat[b] (T,K=2D) * W_dec^T + b_dec) -> f32
  gemm_bf16_wmma<2><<<dim3(D_ / 128, T_ / 64, B_), 256, 0, stream>>>(
      dec_cat, 2 * D_, (long)T_ * 2 * D_, wdec_bf, 2 * D_, 0,
      dwa, D_, (long)T_ * D_, T_, D_, 2 * D_, b_dec);

  // 6) logits[b] (T x V) = dec_cat[b][:, :D] (T,K=D) * W_out^T + b_out -> d_out f32
  gemm_bf16_wmma<1><<<dim3((V_ + 127) / 128, T_ / 64, B_), 256, 0, stream>>>(
      dec_cat, 2 * D_, (long)T_ * 2 * D_, wout_bf, D_, 0,
      out, V_, (long)T_ * V_, T_, V_, D_, b_out);

  // 7) fused p_gen + V-softmax + scatter + log, in place on d_out
  k_finalize<<<dim3(T_, B_), 512, 0, stream>>>(out, dwa, W_gen, b_gen, scores, mids, mseq);
}
